// ResidualBlock_45157286150312
// MI455X (gfx1250) — matrix operand order repaired and checked by reading, not compiled, not run on hardware
//
#include <hip/hip_runtime.h>
#include <hip/hip_bf16.h>
#include <math.h>

typedef __attribute__((ext_vector_type(16))) _Float16 v16h;
typedef __attribute__((ext_vector_type(8)))  _Float16 v8h;
typedef __attribute__((ext_vector_type(4)))  _Float16 v4h;
typedef __attribute__((ext_vector_type(8)))  float    v8f;
typedef __attribute__((ext_vector_type(4)))  float    v4f;

#define NB     64
#define LT     192
#define DD     1024
#define HH     16
#define DHH    64
#define INNERC 1024
#define DFFC   4096
#define LCC    192
#define MLR    (NB * LT)   // 12288 flattened rows

__device__ __forceinline__ v8f wmma16(v16h a, v16h b, v8f c) {
  return __builtin_amdgcn_wmma_f32_16x16x32_f16(false, a, false, b, (short)0, c,
                                                false, false);
}

// ---------------------------------------------------------------------------
// Generic tiled WMMA GEMM:  C[M,N] = act( A[M,K] * op(B) + bias + resid )
//   TRANSB=1: B stored [N,K] row-major (the "@ W.T" case)
//   TRANSB=0: B stored [K,N] row-major
//   ACT: 0 = none, 1 = exact GELU
// Batched over blockIdx.z with element strides (0 => shared operand).
// Software-pipelined: K-tile t+1 global loads overlap K-tile t WMMAs.
// ---------------------------------------------------------------------------
#define BM 128
#define BN 128
#define BK 32
#define LDSS 40   // half stride per LDS row (32 + 8 pad, keeps 16B alignment)

template <bool TRANSB, int ACT>
__global__ __launch_bounds__(256) void gemm_wmma(
    const float* __restrict__ A, const float* __restrict__ Bm,
    const float* __restrict__ bias, const float* __restrict__ resid,
    float* __restrict__ C, int M, int N, int K,
    long strideA, long strideB, long strideC) {
  __shared__ _Float16 As[BM * LDSS];
  __shared__ _Float16 Bs[BN * LDSS];

  const int tid  = threadIdx.x;
  const int lane = tid & 31;
  const int wave = tid >> 5;
  const int wm   = (wave >> 2) * 64;  // 0 / 64
  const int wn   = (wave & 3) * 32;   // 0..96
  const int r16  = lane & 15;
  const int hi   = lane >> 4;         // 0 / 1
  const int kg8  = hi * 8;

  const int mbase = blockIdx.y * BM;
  const int nbase = blockIdx.x * BN;
  const long z = blockIdx.z;
  A += z * strideA;
  Bm += z * strideB;
  C += z * strideC;
  const float* R = resid ? resid + z * strideC : nullptr;

  v8f acc[4][2];
#pragma unroll
  for (int i = 0; i < 4; i++)
#pragma unroll
    for (int j = 0; j < 2; j++)
#pragma unroll
      for (int r = 0; r < 8; r++) acc[i][j][r] = 0.f;

  v4f aReg[4], bReg[4];

  auto load_tile = [&](int kb) {
#pragma unroll
    for (int it = 0; it < 4; it++) {
      int i = tid + it * 256;          // [0, 1024): BM x BK/4
      int r = i >> 3;
      int c4 = (i & 7) * 4;
      int gr = mbase + r;
      v4f val = {0.f, 0.f, 0.f, 0.f};
      if (gr < M) val = *(const v4f*)(A + (long)gr * K + kb + c4);
      aReg[it] = val;
    }
    if (TRANSB) {
#pragma unroll
      for (int it = 0; it < 4; it++) {
        int i = tid + it * 256;
        int n = i >> 3;
        int c4 = (i & 7) * 4;
        int gn = nbase + n;
        v4f val = {0.f, 0.f, 0.f, 0.f};
        if (gn < N) val = *(const v4f*)(Bm + (long)gn * K + kb + c4);
        bReg[it] = val;
      }
    } else {
#pragma unroll
      for (int it = 0; it < 4; it++) {
        int i = tid + it * 256;        // [0, 1024): BK x BN/4
        int k = i >> 5;
        int n4 = (i & 31) * 4;
        int gn = nbase + n4;
        v4f val = {0.f, 0.f, 0.f, 0.f};
        if (gn + 3 < N && (kb + k) < K)
          val = *(const v4f*)(Bm + (long)(kb + k) * N + gn);
        bReg[it] = val;
      }
    }
  };

  auto store_tile = [&]() {
#pragma unroll
    for (int it = 0; it < 4; it++) {
      int i = tid + it * 256;
      int r = i >> 3;
      int c4 = (i & 7) * 4;
      v4h h;
      h[0] = (_Float16)aReg[it][0]; h[1] = (_Float16)aReg[it][1];
      h[2] = (_Float16)aReg[it][2]; h[3] = (_Float16)aReg[it][3];
      *(v4h*)&As[r * LDSS + c4] = h;   // one ds_store_b64
    }
    if (TRANSB) {
#pragma unroll
      for (int it = 0; it < 4; it++) {
        int i = tid + it * 256;
        int n = i >> 3;
        int c4 = (i & 7) * 4;
        v4h h;
        h[0] = (_Float16)bReg[it][0]; h[1] = (_Float16)bReg[it][1];
        h[2] = (_Float16)bReg[it][2]; h[3] = (_Float16)bReg[it][3];
        *(v4h*)&Bs[n * LDSS + c4] = h;
      }
    } else {
#pragma unroll
      for (int it = 0; it < 4; it++) {
        int i = tid + it * 256;
        int k = i >> 5;
        int n4 = (i & 31) * 4;
        Bs[(n4 + 0) * LDSS + k] = (_Float16)bReg[it][0];
        Bs[(n4 + 1) * LDSS + k] = (_Float16)bReg[it][1];
        Bs[(n4 + 2) * LDSS + k] = (_Float16)bReg[it][2];
        Bs[(n4 + 3) * LDSS + k] = (_Float16)bReg[it][3];
      }
    }
  };

  load_tile(0);
  for (int kb = 0; kb < K; kb += BK) {
    store_tile();
    __syncthreads();
    if (kb + BK < K) load_tile(kb + BK);  // overlap next tile with WMMAs
    if (kb + 2 * BK < K) {                // prime L2 two tiles ahead
      __builtin_prefetch(A + (long)(mbase + (tid >> 1)) * K + kb + 2 * BK, 0, 1);
      if (TRANSB)
        __builtin_prefetch(Bm + (long)(nbase + (tid >> 1)) * K + kb + 2 * BK, 0, 1);
      else
        __builtin_prefetch(Bm + (long)(kb + 2 * BK + (tid >> 3)) * N + nbase, 0, 1);
    }

    v16h afr[4], bfr[2];
#pragma unroll
    for (int mt = 0; mt < 4; mt++) {
      // A layout: lane row = r16; halves 0..7 -> K = kg8+i ; 8..15 -> 16+kg8+(i-8)
      const _Float16* p = &As[(wm + mt * 16 + r16) * LDSS];
      v8h lo = *(const v8h*)(p + kg8);
      v8h hb = *(const v8h*)(p + 16 + kg8);
      afr[mt] = __builtin_shufflevector(lo, hb, 0, 1, 2, 3, 4, 5, 6, 7, 8, 9,
                                        10, 11, 12, 13, 14, 15);
    }
#pragma unroll
    for (int nt = 0; nt < 2; nt++) {
      // B layout: lane col = r16; halves 0..7 -> K = kg8+i ; 8..15 -> 16+kg8+(i-8)
      const _Float16* p = &Bs[(wn + nt * 16 + r16) * LDSS];
      v8h lo = *(const v8h*)(p + kg8);
      v8h hb = *(const v8h*)(p + 16 + kg8);
      bfr[nt] = __builtin_shufflevector(lo, hb, 0, 1, 2, 3, 4, 5, 6, 7, 8, 9,
                                        10, 11, 12, 13, 14, 15);
    }
#pragma unroll
    for (int mt = 0; mt < 4; mt++)
#pragma unroll
      for (int nt = 0; nt < 2; nt++)
        acc[mt][nt] = wmma16(afr[mt], bfr[nt], acc[mt][nt]);
    __syncthreads();
  }

  // ---- epilogue: C layout = col per lane, 8 rows per lane group ----
#pragma unroll
  for (int mt = 0; mt < 4; mt++) {
#pragma unroll
    for (int nt = 0; nt < 2; nt++) {
      int col = nbase + wn + nt * 16 + r16;
      if (col >= N) continue;
      float bv = bias ? bias[col] : 0.f;
#pragma unroll
      for (int r = 0; r < 8; r++) {
        int row = mbase + wm + mt * 16 + hi * 8 + r;
        if (row >= M) continue;
        float v = acc[mt][nt][r] + bv;
        if (R) v += R[(long)row * N + col];
        if (ACT == 1) v = 0.5f * v * (1.f + erff(v * 0.70710678118654752f));
        C[(long)row * N + col] = v;
      }
    }
  }
}

// ---------------------------------------------------------------------------
// Fused flash cross-attention: one wave per (qtile, head, batch).
// Computes S^T = K*Q^T so softmax stats live in-lane; P fragment is built
// in-register in exact WMMA-A layout, O accumulated with P*V.
// V is pre-transposed to [B, H, DH, LC] so all fragment loads are b128.
// ---------------------------------------------------------------------------
__global__ __launch_bounds__(32) void attention_kernel(
    const float* __restrict__ q, const float* __restrict__ k,
    const float* __restrict__ vt, float* __restrict__ o) {
  const int qt = blockIdx.x, h = blockIdx.y, b = blockIdx.z;
  const int lane = threadIdx.x;
  const int r16 = lane & 15;
  const int hi = lane >> 4;
  const int kg8 = hi * 8;
  const float scale = 0.125f;  // DH^-0.5

  // Q in WMMA-B layout: lane = query col r16; halves i -> dh = kt*32 + hi*16 + i
  const long qrow = (long)b * LT + qt * 16;
  v16h qf[2];
  {
    const float* qp = q + (qrow + r16) * INNERC + h * DHH;
#pragma unroll
    for (int kt = 0; kt < 2; kt++) {
      const float* p = qp + kt * 32 + hi * 16;   // 16 contiguous floats
      v16h t;
#pragma unroll
      for (int c = 0; c < 4; c++) {
        v4f f = *(const v4f*)(p + c * 4);
#pragma unroll
        for (int j = 0; j < 4; j++) t[c * 4 + j] = (_Float16)(f[j] * scale);
      }
      qf[kt] = t;
    }
  }

  float m_run = -1e30f, s_run = 0.f;
  v8f O[4];
#pragma unroll
  for (int nt = 0; nt < 4; nt++)
#pragma unroll
    for (int r = 0; r < 8; r++) O[nt][r] = 0.f;

  for (int kb = 0; kb < LCC; kb += 32) {
    // S^T sub-tiles: S[s] covers keys kb+s*16 .. kb+s*16+15 (rows), queries (cols)
    v8f S[2];
#pragma unroll
    for (int s = 0; s < 2; s++) {
#pragma unroll
      for (int r = 0; r < 8; r++) S[s][r] = 0.f;
      const float* kp = k + ((long)b * LCC + kb + s * 16 + r16) * INNERC + h * DHH;
#pragma unroll
      for (int kt = 0; kt < 2; kt++) {
        // WMMA-A layout over dh: one contiguous 16-float run (dh = kt*32 + hi*16 + i)
        const float* p0 = kp + kt * 32 + hi * 16;
        const float* p1 = kp + kt * 32 + hi * 16 + 8;
        v4f f0 = *(const v4f*)p0, f1 = *(const v4f*)(p0 + 4);
        v4f f2 = *(const v4f*)p1, f3 = *(const v4f*)(p1 + 4);
        v16h kf;
#pragma unroll
        for (int j = 0; j < 4; j++) {
          kf[j]      = (_Float16)f0[j];
          kf[4 + j]  = (_Float16)f1[j];
          kf[8 + j]  = (_Float16)f2[j];
          kf[12 + j] = (_Float16)f3[j];
        }
        S[s] = wmma16(kf, qf[kt], S[s]);
      }
    }
    // online softmax: lane pair (q, q+16) holds query r16's 32 scores
    float tmax = -1e30f;
#pragma unroll
    for (int r = 0; r < 8; r++) tmax = fmaxf(tmax, fmaxf(S[0][r], S[1][r]));
    tmax = fmaxf(tmax, __shfl_xor(tmax, 16, 32));
    float mnew = fmaxf(m_run, tmax);
    float alpha = __expf(m_run - mnew);
    float p0[8], p1[8], rs = 0.f;
#pragma unroll
    for (int r = 0; r < 8; r++) {
      p0[r] = __expf(S[0][r] - mnew);
      p1[r] = __expf(S[1][r] - mnew);
      rs += p0[r] + p1[r];
    }
    rs += __shfl_xor(rs, 16, 32);
    s_run = s_run * alpha + rs;
    m_run = mnew;
    // rescale O rows (row = query hi*8+r; its alpha lives in lane hi*8+r)
    float af[8];
#pragma unroll
    for (int r = 0; r < 8; r++) af[r] = __shfl(alpha, hi * 8 + r, 32);
#pragma unroll
    for (int nt = 0; nt < 4; nt++)
#pragma unroll
      for (int r = 0; r < 8; r++) O[nt][r] *= af[r];
    // P fragment (WMMA-A layout) directly from in-lane scores
    v16h pf;
#pragma unroll
    for (int r = 0; r < 8; r++) {
      pf[r] = (_Float16)p0[r];
      pf[8 + r] = (_Float16)p1[r];
    }
    // V fragments (WMMA-B layout): lane col = dh nt*16+r16; halves 0..7 -> key kg8+i, 8..15 -> 16+kg8+(i-8)
    // vt layout [B,H,DH,LC] -> two 8-float runs along LC
#pragma unroll
    for (int nt = 0; nt < 4; nt++) {
      const float* vp = vt + (((long)b * HH + h) * DHH + nt * 16 + r16) * LCC +
                        kb + kg8;
      v16h vf;
#pragma unroll
      for (int c = 0; c < 4; c++) {
        v4f f = *(const v4f*)(vp + (c >> 1) * 16 + (c & 1) * 4);
#pragma unroll
        for (int j = 0; j < 4; j++) vf[c * 4 + j] = (_Float16)f[j];
      }
      O[nt] = wmma16(pf, vf, O[nt]);
    }
  }
  // normalize + store
  float sf[8];
#pragma unroll
  for (int r = 0; r < 8; r++) sf[r] = __shfl(s_run, hi * 8 + r, 32);
#pragma unroll
  for (int nt = 0; nt < 4; nt++) {
    int col = h * DHH + nt * 16 + r16;
#pragma unroll
    for (int r = 0; r < 8; r++) {
      long row = qrow + hi * 8 + r;
      o[row * INNERC + col] = O[nt][r] / sf[r];
    }
  }
}

// ---------------------------------------------------------------------------
// Elementwise / reduction kernels
// ---------------------------------------------------------------------------
__global__ __launch_bounds__(256) void layernorm_kernel(
    const float* __restrict__ x, const float* __restrict__ g,
    const float* __restrict__ b, float* __restrict__ out, int D) {
  __shared__ float red[256];
  const long row = blockIdx.x;
  const float* xr = x + row * D;
  float* orow = out + row * D;
  const int tid = threadIdx.x;
  float s = 0.f;
  for (int i = tid; i < D; i += 256) s += xr[i];
  red[tid] = s; __syncthreads();
  for (int off = 128; off > 0; off >>= 1) {
    if (tid < off) red[tid] += red[tid + off];
    __syncthreads();
  }
  float mean = red[0] / D; __syncthreads();
  float vv = 0.f;
  for (int i = tid; i < D; i += 256) { float d = xr[i] - mean; vv += d * d; }
  red[tid] = vv; __syncthreads();
  for (int off = 128; off > 0; off >>= 1) {
    if (tid < off) red[tid] += red[tid + off];
    __syncthreads();
  }
  float rstd = rsqrtf(red[0] / D + 1e-5f); __syncthreads();
  for (int i = tid; i < D; i += 256)
    orow[i] = (xr[i] - mean) * rstd * g[i] + b[i];
}

__global__ __launch_bounds__(256) void softplus_norm_kernel(
    const float* __restrict__ w, float* __restrict__ Aout) {
  __shared__ float red[256];
  __shared__ float vals[LT];
  const int row = blockIdx.x, tid = threadIdx.x;
  float s = 0.f;
  for (int i = tid; i < LT; i += 256) {
    float x = w[row * LT + i];
    float sp = (x > 20.f) ? x : log1pf(__expf(x));
    vals[i] = sp; s += sp;
  }
  red[tid] = s; __syncthreads();
  for (int off = 128; off > 0; off >>= 1) {
    if (tid < off) red[tid] += red[tid + off];
    __syncthreads();
  }
  float inv = 1.f / red[0];
  for (int i = tid; i < LT; i += 256) Aout[row * LT + i] = vals[i] * inv;
}

__global__ __launch_bounds__(256) void add_diff_kernel(
    const float* __restrict__ x, const float* __restrict__ dp,
    float* __restrict__ out, long total) {
  long idx = (long)blockIdx.x * 256 + threadIdx.x;
  if (idx >= total) return;
  long b = idx / ((long)LT * DD);
  int d = (int)(idx % DD);
  out[idx] = x[idx] + dp[b * DD + d];
}

__global__ __launch_bounds__(256) void transpose_v_kernel(
    const float* __restrict__ v, float* __restrict__ vt, long total) {
  long idx = (long)blockIdx.x * 256 + threadIdx.x;
  if (idx >= total) return;
  long b = idx / ((long)LCC * INNERC);
  long rem = idx % ((long)LCC * INNERC);
  int l = (int)(rem / INNERC);
  int c = (int)(rem % INNERC);
  vt[((long)b * INNERC + c) * LCC + l] = v[idx];
}

__global__ __launch_bounds__(256) void gate_kernel(
    const float* __restrict__ x7, float* __restrict__ out, long total) {
  long idx = (long)blockIdx.x * 256 + threadIdx.x;
  if (idx >= total) return;
  long m = idx / 512;
  int j = (int)(idx % 512);
  float g = x7[m * 1024 + j];
  float f = x7[m * 1024 + 512 + j];
  out[idx] = tanhf(g) * (1.f / (1.f + __expf(-f)));
}

__global__ __launch_bounds__(256) void final_kernel(
    const float* __restrict__ outk, const float* __restrict__ res,
    float* __restrict__ dout, long total) {
  long idx = (long)blockIdx.x * 256 + threadIdx.x;
  if (idx >= total) return;
  long m = idx / DD;
  int d = (int)(idx % DD);
  float a = outk[m * 2048 + d];
  a = (a >= 0.f) ? a : 0.4f * a;
  float sk = outk[m * 2048 + 1024 + d];
  sk = (sk >= 0.f) ? sk : 0.4f * sk;
  dout[idx] = (res[idx] + a) * 0.70710678118654752f;
  dout[total + idx] = sk;
}

// ---------------------------------------------------------------------------
static inline dim3 gemm_grid(int M, int N, int batch) {
  return dim3((N + BN - 1) / BN, (M + BM - 1) / BM, batch);
}

extern "C" void kernel_launch(void* const* d_in, const int* in_sizes, int n_in,
                              void* d_out, int out_size, void* d_ws,
                              size_t ws_size, hipStream_t stream) {
  (void)in_sizes; (void)n_in; (void)out_size; (void)ws_size;
  const float* x        = (const float*)d_in[0];
  const float* conds    = (const float*)d_in[1];
  const float* diff_emb = (const float*)d_in[2];
  const float* w_diff   = (const float*)d_in[3];
  const float* b_diff   = (const float*)d_in[4];
  const float* w_cond   = (const float*)d_in[5];
  const float* b_cond   = (const float*)d_in[6];
  const float* wq       = (const float*)d_in[7];
  const float* wk       = (const float*)d_in[8];
  const float* wv       = (const float*)d_in[9];
  const float* wo       = (const float*)d_in[10];
  const float* bo       = (const float*)d_in[11];
  const float* ln1_g    = (const float*)d_in[12];
  const float* ln1_b    = (const float*)d_in[13];
  const float* ln2_g    = (const float*)d_in[14];
  const float* ln2_b    = (const float*)d_in[15];
  const float* w_out    = (const float*)d_in[16];
  const float* b_out    = (const float*)d_in[17];
  const float* enc_wv   = (const float*)d_in[18];
  const float* enc_bv   = (const float*)d_in[19];
  const float* enc_wo   = (const float*)d_in[20];
  const float* enc_bo   = (const float*)d_in[21];
  const float* wmat     = (const float*)d_in[22];
  const float* conv1_w  = (const float*)d_in[23];
  const float* conv1_b  = (const float*)d_in[24];
  const float* conv2_w  = (const float*)d_in[25];
  const float* conv2_b  = (const float*)d_in[26];
  const float* n1_g     = (const float*)d_in[27];
  const float* n1_b     = (const float*)d_in[28];
  const float* n2_g     = (const float*)d_in[29];
  const float* n2_b     = (const float*)d_in[30];
  float* out = (float*)d_out;

  // ---- workspace carve (floats) ----
  float* ws = (float*)d_ws;
  long off = 0;
  const long MD = (long)MLR * DD;
  float* b_x = ws + off; off += MD;            // xbuf -> x4
  float* b_v = ws + off; off += MD;            // values -> x2 -> t -> x5 -> v
  float* b_n = ws + off; off += MD;            // newx -> x3 -> q -> x7
  float* b_k = ws + off; off += MD;            // k -> x6
  float* b_c = ws + off; off += MD;            // cond -> o -> gated
  float* b_t = ws + off; off += MD;            // v transposed [B,H,DH,LC]
  float* b_y = ws + off; off += 2048L * DFFC;  // FFN hidden chunk
  float* b_o = ws + off; off += (long)MLR * 2048L;  // w_out result
  float* b_A = ws + off; off += (long)LT * LT;      // normalized mixing matrix
  float* b_dp = ws + off; off += (long)NB * DD;     // diff projection

  const long elN = MD;
  const int eb = 256;
  const long egrid = (elN + eb - 1) / eb;

  // 1) diffp = diff_emb @ w_diff^T + b_diff   [64,1024]
  gemm_wmma<true, 0><<<gemm_grid(NB, DD, 1), 256, 0, stream>>>(
      diff_emb, w_diff, b_diff, nullptr, b_dp, NB, DD, 256, 0, 0, 0);
  // 2) xbuf = x + diffp broadcast
  add_diff_kernel<<<egrid, eb, 0, stream>>>(x, b_dp, b_x, elN);
  // 3) values = xbuf @ enc_wv^T + enc_bv
  gemm_wmma<true, 0><<<gemm_grid(MLR, DD, 1), 256, 0, stream>>>(
      b_x, enc_wv, enc_bv, nullptr, b_v, MLR, DD, DD, 0, 0, 0);
  // 4) A = rownorm(softplus(weight_mat))
  softplus_norm_kernel<<<LT, 256, 0, stream>>>(wmat, b_A);
  // 5) newx[b] = A @ values[b]   (batched, shared A, B non-transposed)
  gemm_wmma<false, 0><<<gemm_grid(LT, DD, NB), 256, 0, stream>>>(
      b_A, b_v, nullptr, nullptr, b_n, LT, DD, LT, 0, (long)LT * DD,
      (long)LT * DD);
  // 6) x2 = xbuf + newx @ enc_wo^T + enc_bo
  gemm_wmma<true, 0><<<gemm_grid(MLR, DD, 1), 256, 0, stream>>>(
      b_n, enc_wo, enc_bo, b_x, b_v, MLR, DD, DD, 0, 0, 0);
  // 7) x3 = LN(x2, enc_n1)
  layernorm_kernel<<<MLR, 256, 0, stream>>>(b_v, n1_g, n1_b, b_n, DD);
  // 8) FFN chunked over M (hidden never fully materialized)
  for (int c = 0; c < 6; c++) {
    const long ro = (long)c * 2048 * DD;
    gemm_wmma<true, 1><<<gemm_grid(2048, DFFC, 1), 256, 0, stream>>>(
        b_n + ro, conv1_w, conv1_b, nullptr, b_y, 2048, DFFC, DD, 0, 0, 0);
    gemm_wmma<true, 0><<<gemm_grid(2048, DD, 1), 256, 0, stream>>>(
        b_y, conv2_w, conv2_b, b_n + ro, b_v + ro, 2048, DD, DFFC, 0, 0, 0);
  }
  // 9) x4 = LN(x3+y, enc_n2); 10) x5 = LN(x4, ln1)
  layernorm_kernel<<<MLR, 256, 0, stream>>>(b_v, n2_g, n2_b, b_x, DD);
  layernorm_kernel<<<MLR, 256, 0, stream>>>(b_x, ln1_g, ln1_b, b_v, DD);
  // 11) cond = conds @ w_cond^T + b_cond
  gemm_wmma<true, 0><<<gemm_grid(MLR, DD, 1), 256, 0, stream>>>(
      conds, w_cond, b_cond, nullptr, b_c, MLR, DD, 512, 0, 0, 0);
  // 12-14) q,k,v projections
  gemm_wmma<true, 0><<<gemm_grid(MLR, DD, 1), 256, 0, stream>>>(
      b_v, wq, nullptr, nullptr, b_n, MLR, DD, DD, 0, 0, 0);
  gemm_wmma<true, 0><<<gemm_grid(MLR, DD, 1), 256, 0, stream>>>(
      b_c, wk, nullptr, nullptr, b_k, MLR, DD, DD, 0, 0, 0);
  gemm_wmma<true, 0><<<gemm_grid(MLR, DD, 1), 256, 0, stream>>>(
      b_c, wv, nullptr, nullptr, b_v, MLR, DD, DD, 0, 0, 0);
  // 14b) transpose V -> [B,H,DH,LC] for contiguous attention fragments
  transpose_v_kernel<<<egrid, eb, 0, stream>>>(b_v, b_t, elN);
  // 15) fused flash attention -> o
  attention_kernel<<<dim3(LT / 16, HH, NB), 32, 0, stream>>>(b_n, b_k, b_t, b_c);
  // 16) x6 = o @ wo^T + bo ; 17) x7 = LN(x6, ln2)
  gemm_wmma<true, 0><<<gemm_grid(MLR, DD, 1), 256, 0, stream>>>(
      b_c, wo, bo, nullptr, b_k, MLR, DD, DD, 0, 0, 0);
  layernorm_kernel<<<MLR, 256, 0, stream>>>(b_k, ln2_g, ln2_b, b_n, DD);
  // 18) gated = tanh(gate)*sigmoid(filt)
  const long gN = (long)MLR * 512;
  gate_kernel<<<(gN + eb - 1) / eb, eb, 0, stream>>>(b_n, b_c, gN);
  // 19) outk = gated @ w_out^T + b_out   [MLR, 2048]
  gemm_wmma<true, 0><<<gemm_grid(MLR, 2048, 1), 256, 0, stream>>>(
      b_c, w_out, b_out, nullptr, b_o, MLR, 2048, 512, 0, 0, 0);
  // 20) leaky-relu, split, residual, scale -> d_out = [out | skip]
  final_kernel<<<egrid, eb, 0, stream>>>(b_o, x, out, elN);
}
